// GazePatternAttention_79482664779919
// MI455X (gfx1250) — matrix operand order repaired and checked by reading, not compiled, not run on hardware
//
#include <hip/hip_runtime.h>

typedef _Float16 f16;
typedef __attribute__((ext_vector_type(16))) _Float16 v16h;
typedef __attribute__((ext_vector_type(8)))  _Float16 v8h;
typedef __attribute__((ext_vector_type(8)))  float    v8f;

union AFrag { v16h v; v8h h[2]; };

#define DIMD   256
#define HEADS  4
#define HD     64
#define SEQ    512
#define BATCH  32
#define MROWS  (BATCH * SEQ)
#define SCALEA 0.125f        // HD^-0.5
#define WIN    10
#define PADW   5

// ---------------------------------------------------------------- conversion
__global__ void cvt_f32_f16(const float* __restrict__ in, f16* __restrict__ out, int n) {
  int i = blockIdx.x * blockDim.x + threadIdx.x;
  if (i < n) out[i] = (f16)in[i];
}

// ---------------------------------------------------------------- weight pack
// Pack f32 weight W (K x N, row-major) into WMMA B-fragment layout:
//   Wp[(nt*ktiles + kc)*512 + lane*16 + e] = W[kc*32 + 16*(lane>=16) + e][nt*16 + lane%16]
// so a B fragment is 2 contiguous b128 loads per lane, and the GEMM k-loop
// streams 1KB fragments sequentially.
__global__ void pack_w(const float* __restrict__ W, f16* __restrict__ Wp, int K, int N) {
  const int tid = blockIdx.x * blockDim.x + threadIdx.x;
  const int lane = tid & 31;
  const int f = tid >> 5;
  const int ktiles = K >> 5, ntiles = N >> 4;
  if (f >= ktiles * ntiles) return;
  const int nt = f / ktiles, kc = f % ktiles;
  const int hi = lane >> 4, l16 = lane & 15;
  const float* src = W + (size_t)(kc * 32 + 16 * hi) * N + nt * 16 + l16;
  f16* dst = Wp + (size_t)f * 512 + lane * 16;
  #pragma unroll
  for (int e = 0; e < 16; ++e) dst[e] = (f16)src[(size_t)e * N];
}

// ---------------------------------------------------------------- V pack
// Pack V (f16, [B*SEQ, DIMD]) into B-fragment layout for P@V:
//   frag g = ((b*HEADS + h)*4 + dt)*16 + sc   (sc = s-chunk of 32, dt = d-tile of 16)
__global__ void pack_v(const f16* __restrict__ V, f16* __restrict__ Vp) {
  const int tid = blockIdx.x * blockDim.x + threadIdx.x;
  const int lane = tid & 31;
  int f = tid >> 5;
  if (f >= BATCH * HEADS * 4 * 16) return;
  const int sc = f & 15; int r = f >> 4;
  const int dt = r & 3; r >>= 2;
  const int h = r & 3; const int b = r >> 2;
  const int hi = lane >> 4, l16 = lane & 15;
  const f16* src = V + (size_t)(b * SEQ + sc * 32 + 16 * hi) * DIMD + h * HD + dt * 16 + l16;
  f16* dst = Vp + (size_t)f * 512 + lane * 16;
  #pragma unroll
  for (int e = 0; e < 16; ++e) dst[e] = src[(size_t)e * DIMD];
}

// ---------------------------------------------------------------- WMMA GEMM
// Y[m, coff+n] = act( A[m,0:K] @ W[0:K, n] + bias[n] )  (+ optional f32 residual)
// A: f16 row-major. Wp: packed B fragments (ktiles = K/32). N is always 256 here.
// Each wave computes a 32x32 tile (2x2 WMMA tiles); block (32,4) -> 64x64 tile.
__global__ void __launch_bounds__(128) gemm_wmma(
    const f16* __restrict__ A, int lda,
    const f16* __restrict__ Wp,
    const float* __restrict__ bias,
    f16* __restrict__ Yh, float* __restrict__ Yf, int ldy, int coff,
    const float* __restrict__ resid, int K, int relu)
{
  const int lane = threadIdx.x;
  const int wave = threadIdx.y;                       // 0..3
  const int hi = lane >> 4, l16 = lane & 15;
  const int m0 = blockIdx.x * 64 + (wave >> 1) * 32;
  const int n0 = blockIdx.y * 64 + (wave & 1) * 32;
  const int ktiles = K >> 5;

  v8f acc[2][2];
  #pragma unroll
  for (int bt = 0; bt < 2; ++bt) {
    const float b0 = bias[n0 + bt * 16 + l16];
    #pragma unroll
    for (int at = 0; at < 2; ++at)
      #pragma unroll
      for (int j = 0; j < 8; ++j) acc[at][bt][j] = b0;
  }

  const f16* arow0 = A + (size_t)(m0 + l16) * lda;
  const f16* arow1 = A + (size_t)(m0 + 16 + l16) * lda;
  const int abase = hi * 16;
  const f16* bp0 = Wp + ((size_t)((n0 >> 4) + 0) * ktiles) * 512 + lane * 16;
  const f16* bp1 = Wp + ((size_t)((n0 >> 4) + 1) * ktiles) * 512 + lane * 16;

  for (int k0 = 0; k0 < K; k0 += 32) {
    AFrag a0, a1, b0f, b1f;
    a0.h[0] = *(const v8h*)(arow0 + k0 + abase);
    a0.h[1] = *(const v8h*)(arow0 + k0 + 8 + abase);
    a1.h[0] = *(const v8h*)(arow1 + k0 + abase);
    a1.h[1] = *(const v8h*)(arow1 + k0 + 8 + abase);
    const size_t fo = (size_t)(k0 >> 5) * 512;
    b0f.h[0] = *(const v8h*)(bp0 + fo);
    b0f.h[1] = *(const v8h*)(bp0 + fo + 8);
    b1f.h[0] = *(const v8h*)(bp1 + fo);
    b1f.h[1] = *(const v8h*)(bp1 + fo + 8);
    acc[0][0] = __builtin_amdgcn_wmma_f32_16x16x32_f16(false, a0.v, false, b0f.v, (short)0, acc[0][0], false, false);
    acc[0][1] = __builtin_amdgcn_wmma_f32_16x16x32_f16(false, a0.v, false, b1f.v, (short)0, acc[0][1], false, false);
    acc[1][0] = __builtin_amdgcn_wmma_f32_16x16x32_f16(false, a1.v, false, b0f.v, (short)0, acc[1][0], false, false);
    acc[1][1] = __builtin_amdgcn_wmma_f32_16x16x32_f16(false, a1.v, false, b1f.v, (short)0, acc[1][1], false, false);
  }

  #pragma unroll
  for (int at = 0; at < 2; ++at)
    #pragma unroll
    for (int bt = 0; bt < 2; ++bt)
      #pragma unroll
      for (int j = 0; j < 8; ++j) {
        const int m = m0 + at * 16 + j + 8 * hi;
        const int n = coff + n0 + bt * 16 + l16;
        float val = acc[at][bt][j];
        if (relu) val = fmaxf(val, 0.0f);
        if (Yh) Yh[(size_t)m * ldy + n] = (f16)val;
        if (Yf) {
          float r = resid ? resid[(size_t)m * ldy + n] : 0.0f;
          Yf[(size_t)m * ldy + n] = val + r;
        }
      }
}

// ------------------------------------------------------- global attention
// One block per (b, h, 16-row t tile). 8 waves -> full 16x512 score matrix by
// WMMA QK^T; block softmax in LDS; 4 waves -> P@V by WMMA with packed V.
__global__ void __launch_bounds__(256) attn_global(
    const f16* __restrict__ qh, const f16* __restrict__ kh, const f16* __restrict__ vp,
    const float* __restrict__ rel, f16* __restrict__ gout)
{
  const int tileT = blockIdx.x, h = blockIdx.y, b = blockIdx.z;
  const int tid = threadIdx.x, lane = tid & 31, wave = tid >> 5;
  const int hi = lane >> 4, l16 = lane & 15;
  const int t0 = tileT * 16;

  __shared__ float sc[16][SEQ];
  __shared__ f16   pl[16][SEQ];
  __shared__ float red[16][17];
  __shared__ float rowstat[16];

  // Q fragments (row = t0+l16), kk chunks {0,32} of HD=64
  AFrag aq[2];
  {
    const f16* qrow = qh + (size_t)(b * SEQ + t0 + l16) * DIMD + h * HD;
    #pragma unroll
    for (int c = 0; c < 2; ++c) {
      const int kk = c * 32, base = hi * 16;
      aq[c].h[0] = *(const v8h*)(qrow + kk + base);
      aq[c].h[1] = *(const v8h*)(qrow + kk + 8 + base);
    }
  }
  // scores: each wave computes 4 of the 32 S-tiles
  for (int i = 0; i < 4; ++i) {
    const int S0 = (wave * 4 + i) * 16;
    v8f c;
    #pragma unroll
    for (int j = 0; j < 8; ++j) c[j] = 0.0f;
    // B = K^T chunk: N = key s = S0+l16, K(e) = d = kk + e + 16*hi (contiguous)
    const f16* krow = kh + (size_t)(b * SEQ + S0 + l16) * DIMD + h * HD + 16 * hi;
    #pragma unroll
    for (int c2 = 0; c2 < 2; ++c2) {
      AFrag bb;
      bb.h[0] = *(const v8h*)(krow + c2 * 32);
      bb.h[1] = *(const v8h*)(krow + c2 * 32 + 8);
      c = __builtin_amdgcn_wmma_f32_16x16x32_f16(false, aq[c2].v, false, bb.v, (short)0, c, false, false);
    }
    const int s = S0 + l16;
    #pragma unroll
    for (int j = 0; j < 8; ++j) {
      const int m = j + 8 * hi;
      const int t = t0 + m;
      sc[m][s] = c[j] * SCALEA + rel[(size_t)(s - t + SEQ - 1) * HEADS + h];
    }
  }
  __syncthreads();

  // softmax over 512 keys; 16 threads per row
  const int r = tid >> 4, g = tid & 15;
  float mx = -1e30f;
  for (int j = g; j < SEQ; j += 16) mx = fmaxf(mx, sc[r][j]);
  red[r][g] = mx;
  __syncthreads();
  if (g == 0) {
    float m2 = red[r][0];
    for (int q = 1; q < 16; ++q) m2 = fmaxf(m2, red[r][q]);
    rowstat[r] = m2;
  }
  __syncthreads();
  const float rm = rowstat[r];
  float sum = 0.0f;
  for (int j = g; j < SEQ; j += 16) {
    float e = __expf(sc[r][j] - rm);
    sc[r][j] = e;
    sum += e;
  }
  red[r][g] = sum;
  __syncthreads();
  if (g == 0) {
    float s2 = 0.0f;
    for (int q = 0; q < 16; ++q) s2 += red[r][q];
    rowstat[r] = s2;
  }
  __syncthreads();
  const float inv = 1.0f / rowstat[r];
  for (int j = g; j < SEQ; j += 16) pl[r][j] = (f16)(sc[r][j] * inv);
  __syncthreads();

  // P @ V : 4 waves, one 16x16 d-tile each; packed V fragments (b128 loads)
  if (wave < 4) {
    v8f acc;
    #pragma unroll
    for (int j = 0; j < 8; ++j) acc[j] = 0.0f;
    const f16* vpb = vp + ((size_t)((b * HEADS + h) * 4 + wave) * 16) * 512 + lane * 16;
    for (int s0 = 0; s0 < SEQ; s0 += 32) {
      AFrag a, bb;
      const int base = hi * 16;
      a.h[0] = *(const v8h*)&pl[l16][s0 + base];
      a.h[1] = *(const v8h*)&pl[l16][s0 + 8 + base];
      const size_t fo = (size_t)(s0 >> 5) * 512;
      bb.h[0] = *(const v8h*)(vpb + fo);
      bb.h[1] = *(const v8h*)(vpb + fo + 8);
      acc = __builtin_amdgcn_wmma_f32_16x16x32_f16(false, a.v, false, bb.v, (short)0, acc, false, false);
    }
    #pragma unroll
    for (int j = 0; j < 8; ++j) {
      const int m = j + 8 * hi;
      gout[(size_t)(b * SEQ + t0 + m) * DIMD + h * HD + wave * 16 + l16] = (f16)acc[j];
    }
  }
}

// ------------------------------------------------------- local window attention
// Zero-padded rows of x mean lk/lv at padded positions equal their biases.
__global__ void __launch_bounds__(256) attn_local(
    const f16* __restrict__ lq, const f16* __restrict__ lk, const f16* __restrict__ lv,
    const float* __restrict__ lkb, const float* __restrict__ lvb,
    f16* __restrict__ concat)
{
  const int idx = blockIdx.x * blockDim.x + threadIdx.x;
  if (idx >= MROWS) return;
  const int b = idx / SEQ, t = idx % SEQ;
  const f16* qrow = lq + (size_t)idx * DIMD;

  float p[WIN];
  float mx = -1e30f;
  for (int wi = 0; wi < WIN; ++wi) {
    const int sp = t + wi - PADW;
    float s = 0.0f;
    if (sp >= 0 && sp < SEQ) {
      const f16* kr = lk + (size_t)(b * SEQ + sp) * DIMD;
      for (int d0 = 0; d0 < DIMD; d0 += 8) {
        const v8h qv = *(const v8h*)(qrow + d0);
        const v8h kv = *(const v8h*)(kr + d0);
        #pragma unroll
        for (int e = 0; e < 8; ++e) s += (float)qv[e] * (float)kv[e];
      }
    } else {
      for (int d0 = 0; d0 < DIMD; d0 += 8) {
        const v8h qv = *(const v8h*)(qrow + d0);
        #pragma unroll
        for (int e = 0; e < 8; ++e) s += (float)qv[e] * lkb[d0 + e];
      }
    }
    p[wi] = s * SCALEA;
    mx = fmaxf(mx, p[wi]);
  }
  float sum = 0.0f;
  for (int wi = 0; wi < WIN; ++wi) { p[wi] = __expf(p[wi] - mx); sum += p[wi]; }
  const float inv = 1.0f / sum;
  for (int wi = 0; wi < WIN; ++wi) p[wi] *= inv;

  f16* orow = concat + (size_t)idx * (2 * DIMD) + DIMD;
  for (int d0 = 0; d0 < DIMD; d0 += 8) {
    float o[8];
    #pragma unroll
    for (int e = 0; e < 8; ++e) o[e] = 0.0f;
    for (int wi = 0; wi < WIN; ++wi) {
      const int sp = t + wi - PADW;
      if (sp >= 0 && sp < SEQ) {
        const v8h vv = *(const v8h*)(lv + (size_t)(b * SEQ + sp) * DIMD + d0);
        #pragma unroll
        for (int e = 0; e < 8; ++e) o[e] += p[wi] * (float)vv[e];
      } else {
        #pragma unroll
        for (int e = 0; e < 8; ++e) o[e] += p[wi] * lvb[d0 + e];
      }
    }
    v8h ov;
    #pragma unroll
    for (int e = 0; e < 8; ++e) ov[e] = (f16)o[e];
    *(v8h*)(orow + d0) = ov;
  }
}

// ------------------------------------------------------- pattern scores
__global__ void __launch_bounds__(256) pattern_kernel(
    const float* __restrict__ x, const float* __restrict__ pe,
    const float* __restrict__ pqw, const float* __restrict__ pqb,
    const float* __restrict__ pkw, const float* __restrict__ pkb,
    float* __restrict__ out)
{
  const int b = blockIdx.x, tid = threadIdx.x;
  __shared__ float pooled[DIMD], pqv[DIMD], pkv[3][DIMD], lg[HEADS][3];

  {
    float s = 0.0f;
    const float* xb = x + (size_t)b * SEQ * DIMD + tid;
    for (int t = 0; t < SEQ; ++t) s += xb[(size_t)t * DIMD];
    pooled[tid] = s * (1.0f / SEQ);
  }
  __syncthreads();
  {
    float s = pqb[tid];
    for (int k = 0; k < DIMD; ++k) s += pooled[k] * pqw[(size_t)k * DIMD + tid];
    pqv[tid] = s;
    for (int kk = 0; kk < 3; ++kk) {
      float s2 = pkb[tid];
      for (int k = 0; k < DIMD; ++k) s2 += pe[kk * DIMD + k] * pkw[(size_t)k * DIMD + tid];
      pkv[kk][tid] = s2;
    }
  }
  __syncthreads();
  if (tid < HEADS * 3) {
    const int h = tid / 3, kk = tid % 3;
    float s = 0.0f;
    for (int d = 0; d < HD; ++d) s += pqv[h * HD + d] * pkv[kk][h * HD + d];
    lg[h][kk] = s * SCALEA;
  }
  __syncthreads();
  if (tid == 0) {
    float pw[3] = {0.0f, 0.0f, 0.0f};
    for (int h = 0; h < HEADS; ++h) {
      const float m = fmaxf(lg[h][0], fmaxf(lg[h][1], lg[h][2]));
      const float e0 = __expf(lg[h][0] - m);
      const float e1 = __expf(lg[h][1] - m);
      const float e2 = __expf(lg[h][2] - m);
      const float inv = 1.0f / (e0 + e1 + e2);
      pw[0] += e0 * inv; pw[1] += e1 * inv; pw[2] += e2 * inv;
    }
    out[b * 3 + 0] = pw[0] * 0.25f;
    out[b * 3 + 1] = pw[1] * 0.25f;
    out[b * 3 + 2] = pw[2] * 0.25f;
  }
}

// ---------------------------------------------------------------- launch
extern "C" void kernel_launch(void* const* d_in, const int* in_sizes, int n_in,
                              void* d_out, int out_size, void* d_ws, size_t ws_size,
                              hipStream_t stream)
{
  (void)in_sizes; (void)n_in; (void)out_size; (void)ws_size;

  const float* x     = (const float*)d_in[0];
  const float* gq_w  = (const float*)d_in[1];
  const float* gq_b  = (const float*)d_in[2];
  const float* gk_w  = (const float*)d_in[3];
  const float* gk_b  = (const float*)d_in[4];
  const float* gv_w  = (const float*)d_in[5];
  const float* gv_b  = (const float*)d_in[6];
  const float* go_w  = (const float*)d_in[7];
  const float* go_b  = (const float*)d_in[8];
  const float* rel   = (const float*)d_in[9];
  const float* lq_w  = (const float*)d_in[10];
  const float* lq_b  = (const float*)d_in[11];
  const float* lk_w  = (const float*)d_in[12];
  const float* lk_b  = (const float*)d_in[13];
  const float* lv_w  = (const float*)d_in[14];
  const float* lv_b  = (const float*)d_in[15];
  const float* pe    = (const float*)d_in[16];
  const float* pq_w  = (const float*)d_in[17];
  const float* pq_b  = (const float*)d_in[18];
  const float* pk_w  = (const float*)d_in[19];
  const float* pk_b  = (const float*)d_in[20];
  const float* f1_w  = (const float*)d_in[21];
  const float* f1_b  = (const float*)d_in[22];
  const float* f2_w  = (const float*)d_in[23];
  const float* f2_b  = (const float*)d_in[24];

  char* ws = (char*)d_ws;
  size_t off = 0;
  auto alloc_h = [&](size_t n) -> f16* {
    f16* p = (f16*)(ws + off);
    off += ((n * sizeof(f16)) + 255) & ~(size_t)255;
    return p;
  };

  f16* xh   = alloc_h((size_t)MROWS * DIMD);
  f16* wq   = alloc_h(DIMD * DIMD);
  f16* wk   = alloc_h(DIMD * DIMD);
  f16* wv   = alloc_h(DIMD * DIMD);
  f16* wo   = alloc_h(DIMD * DIMD);
  f16* wlq  = alloc_h(DIMD * DIMD);
  f16* wlk  = alloc_h(DIMD * DIMD);
  f16* wlv  = alloc_h(DIMD * DIMD);
  f16* wf1  = alloc_h(2 * DIMD * DIMD);
  f16* wf2  = alloc_h(DIMD * DIMD);
  f16* qh   = alloc_h((size_t)MROWS * DIMD);
  f16* khh  = alloc_h((size_t)MROWS * DIMD);
  f16* vhh  = alloc_h((size_t)MROWS * DIMD);
  f16* lqh  = alloc_h((size_t)MROWS * DIMD);
  f16* lkh  = alloc_h((size_t)MROWS * DIMD);
  f16* lvh  = alloc_h((size_t)MROWS * DIMD);
  f16* vpk  = alloc_h((size_t)MROWS * DIMD);      // packed V fragments
  f16* gouth   = alloc_h((size_t)MROWS * DIMD);
  f16* concath = alloc_h((size_t)MROWS * 2 * DIMD);
  f16* h1h     = alloc_h((size_t)MROWS * DIMD);

  // x -> f16
  {
    const int n = MROWS * DIMD;
    cvt_f32_f16<<<dim3((n + 255) / 256), dim3(256), 0, stream>>>(x, xh, n);
  }
  // weights -> packed B-fragment layout (convert+transpose fused)
  auto pack = [&](const float* s, f16* d, int K, int N) {
    const int threads = (K >> 5) * (N >> 4) * 32;
    pack_w<<<dim3((threads + 255) / 256), dim3(256), 0, stream>>>(s, d, K, N);
  };
  pack(gq_w, wq,  DIMD, DIMD);
  pack(gk_w, wk,  DIMD, DIMD);
  pack(gv_w, wv,  DIMD, DIMD);
  pack(go_w, wo,  DIMD, DIMD);
  pack(lq_w, wlq, DIMD, DIMD);
  pack(lk_w, wlk, DIMD, DIMD);
  pack(lv_w, wlv, DIMD, DIMD);
  pack(f1_w, wf1, 2 * DIMD, DIMD);
  pack(f2_w, wf2, DIMD, DIMD);

  const dim3 gblk(32, 4, 1);
  const dim3 ggrd(MROWS / 64, DIMD / 64, 1);      // (256, 4)

  // projections: q, k, v, lq, lk, lv
  gemm_wmma<<<ggrd, gblk, 0, stream>>>(xh, DIMD, wq,  gq_b, qh,  nullptr, DIMD, 0, nullptr, DIMD, 0);
  gemm_wmma<<<ggrd, gblk, 0, stream>>>(xh, DIMD, wk,  gk_b, khh, nullptr, DIMD, 0, nullptr, DIMD, 0);
  gemm_wmma<<<ggrd, gblk, 0, stream>>>(xh, DIMD, wv,  gv_b, vhh, nullptr, DIMD, 0, nullptr, DIMD, 0);
  gemm_wmma<<<ggrd, gblk, 0, stream>>>(xh, DIMD, wlq, lq_b, lqh, nullptr, DIMD, 0, nullptr, DIMD, 0);
  gemm_wmma<<<ggrd, gblk, 0, stream>>>(xh, DIMD, wlk, lk_b, lkh, nullptr, DIMD, 0, nullptr, DIMD, 0);
  gemm_wmma<<<ggrd, gblk, 0, stream>>>(xh, DIMD, wlv, lv_b, lvh, nullptr, DIMD, 0, nullptr, DIMD, 0);

  // pack V into P@V fragment layout
  {
    const int threads = BATCH * HEADS * 4 * 16 * 32;
    pack_v<<<dim3((threads + 255) / 256), dim3(256), 0, stream>>>(vhh, vpk);
  }

  // global attention -> g_out (f16)
  attn_global<<<dim3(SEQ / 16, HEADS, BATCH), dim3(256), 0, stream>>>(qh, khh, vpk, rel, gouth);

  // go projection into concat[:, 0:256]
  gemm_wmma<<<ggrd, gblk, 0, stream>>>(gouth, DIMD, wo, go_b, concath, nullptr, 2 * DIMD, 0, nullptr, DIMD, 0);

  // local attention into concat[:, 256:512]
  attn_local<<<dim3(MROWS / 256), dim3(256), 0, stream>>>(lqh, lkh, lvh, lk_b, lv_b, concath);

  // f1 (K=512) + ReLU -> h1
  gemm_wmma<<<ggrd, gblk, 0, stream>>>(concath, 2 * DIMD, wf1, f1_b, h1h, nullptr, DIMD, 0, nullptr, 2 * DIMD, 1);

  // f2 + bias + residual x -> d_out (f32)
  gemm_wmma<<<ggrd, gblk, 0, stream>>>(h1h, DIMD, wf2, f2_b, nullptr, (float*)d_out, DIMD, 0, x, DIMD, 0);

  // pattern scores -> tail of d_out
  pattern_kernel<<<dim3(BATCH), dim3(256), 0, stream>>>(
      x, pe, pq_w, pq_b, pk_w, pk_b, (float*)d_out + (size_t)MROWS * DIMD);
}
